// UNIF_59004260712674
// MI455X (gfx1250) — matrix operand order repaired and checked by reading, not compiled, not run on hardware
//
#include <hip/hip_runtime.h>
#include <hip/hip_bf16.h>

typedef _Float16 v16h __attribute__((ext_vector_type(16)));
typedef _Float16 v8h  __attribute__((ext_vector_type(8)));
typedef float    v8f  __attribute__((ext_vector_type(8)));

#define NJ   24
#define NV   20000
#define NG   2
#define DCIN 288
#define HID  128
#define NTILES 1250            // NV / 16
#define WAVES_PER_BLK 8
#define TILES_PER_WAVE 8
#define TILES_PER_BLK (WAVES_PER_BLK * TILES_PER_WAVE)   // 64
#define EPSN 1e-8f

// LDS layout (halves): W0t[128*32] | W1t[128*128] | W2t[128*128] | W3t[16*128] | act[8][16*128]
#define SMEM_HALVES (128*32 + 128*128 + 128*128 + 16*128 + WAVES_PER_BLK*16*128)
#define SMEM_BYTES  (SMEM_HALVES * 2)   // 110592 B

__device__ __forceinline__ float softplus_b(float x) {
    // softplus(100*x)/100 via hardware v_exp_f32 / v_log_f32
    float t = 100.f * x;
    return (fmaxf(t, 0.f) + __logf(1.f + __expf(-fabsf(t)))) * 0.01f;
}

__device__ __forceinline__ float ntn(float x) {
    if (isnan(x)) return 0.f;
    if (isinf(x)) return x > 0.f ? 3.402823466e38f : -3.402823466e38f;
    return x;
}

// ---------------------------------------------------------------------------
// Kernel 1: per-part cond MLP (c is constant over V!) -> cws[j*4 + o]
// ---------------------------------------------------------------------------
__global__ void cond_kernel(const float* __restrict__ Bcond,
                            const float* __restrict__ cW0, const float* __restrict__ cb0,
                            const float* __restrict__ cW1, const float* __restrict__ cb1,
                            const float* __restrict__ cW2, const float* __restrict__ cb2,
                            float* __restrict__ cws) {
    int j = blockIdx.x * blockDim.x + threadIdx.x;
    if (j >= NJ) return;
    float h0[16], h1[16];
    for (int n = 0; n < 16; ++n) {
        float s = cb0[j * 16 + n];
        for (int d = 0; d < DCIN; ++d)
            s += Bcond[j * DCIN + d] * cW0[(j * DCIN + d) * 16 + n];
        h0[n] = softplus_b(s);
    }
    for (int n = 0; n < 16; ++n) {
        float s = cb1[j * 16 + n];
        for (int d = 0; d < 16; ++d)
            s += h0[d] * cW1[(j * 16 + d) * 16 + n];
        h1[n] = softplus_b(s);
    }
    for (int o = 0; o < 4; ++o) {
        float s = cb2[j * 4 + o];
        for (int d = 0; d < 16; ++d)
            s += h1[d] * cW2[(j * 16 + d) * 4 + o];
        cws[j * 4 + o] = s;
    }
}

// ---------------------------------------------------------------------------
// WMMA fragment helpers (LDS staging, f16)
// ---------------------------------------------------------------------------
// A (16x32 f16): lane L: M = L&15; lanes 0-15 hold K 0..15, lanes 16-31
// hold K 16..31 (two contiguous 8-half chunks). act row stride = 128.
__device__ __forceinline__ v16h load_a_frag(const _Float16* act, int lane, int kbase) {
    int M   = lane & 15;
    int klo = (lane < 16) ? 0 : 16;
    const _Float16* p = act + M * 128 + kbase;
    union { v16h v; v8h h[2]; } a;
    a.h[0] = *(const v8h*)(p + klo);
    a.h[1] = *(const v8h*)(p + klo + 8);
    return a.v;
}

// B (32x16 f16): lane L: N = nbase + (L&15); lanes 0-15 hold K kbase+0..15,
// lanes 16-31 hold K kbase+16..31 => 16 contiguous halves of transposed row Wt[N][k].
__device__ __forceinline__ v16h load_b_frag(const _Float16* wt, int kdim,
                                            int lane, int nbase, int kbase) {
    int N    = nbase + (lane & 15);
    int koff = (lane < 16) ? 0 : 16;
    return *(const v16h*)(wt + N * kdim + kbase + koff);
}

// One hidden layer: act(16xKDIM f16 in LDS) x Wt(128xKDIM f16 in LDS) -> act(16x128)
template <int KDIM, int KSTEPS, bool SOFTP>
__device__ __forceinline__ void gemm_layer(_Float16* act, const _Float16* wt,
                                           const float* __restrict__ bias, int lane) {
    v16h af[KSTEPS];
#pragma unroll
    for (int kk = 0; kk < KSTEPS; ++kk) af[kk] = load_a_frag(act, lane, kk * 32);
#pragma unroll
    for (int nt = 0; nt < 8; ++nt) {
        v8f acc = {};
#pragma unroll
        for (int kk = 0; kk < KSTEPS; ++kk) {
            v16h bf = load_b_frag(wt, KDIM, lane, nt * 16, kk * 32);
            acc = __builtin_amdgcn_wmma_f32_16x16x32_f16(
                false, af[kk], false, bf, (short)0, acc, false, false);
        }
        int   N  = nt * 16 + (lane & 15);
        float bv = bias[N];
#pragma unroll
        for (int g = 0; g < 8; ++g) {
            int   M = (lane < 16) ? g : 8 + g;
            float h = acc[g] + bv;
            if (SOFTP) h = softplus_b(h);
            act[M * 128 + N] = (_Float16)h;
        }
    }
}

// ---------------------------------------------------------------------------
// Kernel 2: geometry + grouped MLP via WMMA.
// grid = (ceil(1250/64), 24), block = 256, all weights resident in 108KB LDS,
// each wave loops over 8 M-tiles to amortize the weight staging.
// ---------------------------------------------------------------------------
__global__ __launch_bounds__(256) void
parts_kernel(const float* __restrict__ pts,  const float* __restrict__ Btr,
             const float* __restrict__ Brot, const float* __restrict__ Bneigh,
             const float* __restrict__ ralpha, const float* __restrict__ rbeta,
             const float* __restrict__ W0, const float* __restrict__ b0,
             const float* __restrict__ W1, const float* __restrict__ b1,
             const float* __restrict__ W2, const float* __restrict__ b2,
             const float* __restrict__ W3, const float* __restrict__ b3,
             const float* __restrict__ cws, float* __restrict__ predws) {
    extern __shared__ __align__(32) _Float16 smem[];
    _Float16* s_w0t = smem;                      // [128][32]  Wt0[n][k], k padded 7->32
    _Float16* s_w1t = s_w0t + 128 * 32;          // [128][128] Wt1[n][k]
    _Float16* s_w2t = s_w1t + 128 * 128;         // [128][128] Wt2[n][k]
    _Float16* s_w3t = s_w2t + 128 * 128;         // [16][128]  Wt3[n][k], n padded 1->16
    _Float16* s_actb = s_w3t + 16 * 128;         // [8][16*128]

    const int j    = blockIdx.y;
    const int tid  = threadIdx.x;
    const int wave = tid >> 5;
    const int lane = tid & 31;

    // ---- stage all weights (transposed, f16) once per block ------------------
    for (int idx = tid; idx < 128 * 32; idx += 256) {
        int n = idx >> 5, k = idx & 31;
        s_w0t[idx] = (k < 7) ? (_Float16)W0[j * 7 * 128 + k * 128 + n] : (_Float16)0.f;
    }
    for (int idx = tid; idx < 16 * 128; idx += 256) {
        int n = idx >> 7, k = idx & 127;
        s_w3t[idx] = (n == 0) ? (_Float16)W3[j * 128 + k] : (_Float16)0.f;
    }
    for (int idx = tid; idx < 128 * 128; idx += 256) {
        int n = idx >> 7, k = idx & 127;
        s_w1t[idx] = (_Float16)W1[j * 128 * 128 + k * 128 + n];
        s_w2t[idx] = (_Float16)W2[j * 128 * 128 + k * 128 + n];
    }
    __syncthreads();

    _Float16* act = s_actb + wave * (16 * 128);
    const float* b0g = b0 + j * HID;
    const float* b1g = b1 + j * HID;
    const float* b2g = b2 + j * HID;
    const float  b3v = b3[j];
    const float* cj  = cws + j * 4;

    for (int t = 0; t < TILES_PER_WAVE; ++t) {
        const int tile = blockIdx.x * TILES_PER_BLK + t * WAVES_PER_BLK + wave;
        if (tile >= NTILES) continue;

        // ------------------ geometry prologue: lanes 0..15, one point each ----
        if (lane < 16) {
            int v = tile * 16 + lane;
            const float* P  = pts  + v * 3;
            const float* tr = Btr  + j * 3;
            const float* Rw = Brot + j * 9;
            float qx = P[0] - tr[0], qy = P[1] - tr[1], qz = P[2] - tr[2];
            // pts_local = Brot^T * q
            float lx = Rw[0] * qx + Rw[3] * qy + Rw[6] * qz;
            float ly = Rw[1] * qx + Rw[4] * qy + Rw[7] * qz;
            float lz = Rw[2] * qx + Rw[5] * qy + Rw[8] * qz;

            float offx = 0.f, offy = 0.f, offz = 0.f;
#pragma unroll
            for (int gi = 0; gi < NG; ++gi) {
                const float* nb = Bneigh + ((j * NG + gi) * 5) * 3;
                float jsx = nb[0],  jsy = nb[1],  jsz = nb[2];   // Jtr_this (self)
                float jcx = nb[3],  jcy = nb[4],  jcz = nb[5];   // Jtr_center
                float jnx = nb[6],  jny = nb[7],  jnz = nb[8];   // Jtr_other (neigh)
                float jpx = nb[9],  jpy = nb[10], jpz = nb[11];  // Jpose
                int bt = (int)nb[12];                            // bid_this
                int bo = (int)nb[13];                            // bid_other
                // ra = elu(x-1)+1
                float an_r = ralpha[bo * NJ + bt];
                float as_r = ralpha[bt * NJ + bo];
                float a_n  = (an_r - 1.f > 0.f) ? an_r : __expf(an_r - 1.f);
                float a_s  = (as_r - 1.f > 0.f) ? as_r : __expf(as_r - 1.f);
                float be_n = rbeta[bo * NJ + bt];
                float be_s = rbeta[bt * NJ + bo];
                // bone vectors
                float bnx = jnx - jcx, bny = jny - jcy, bnz = jnz - jcz;
                float bsx = jsx - jcx, bsy = jsy - jcy, bsz = jsz - jcz;
                float nb_n = sqrtf(bnx * bnx + bny * bny + bnz * bnz);
                float nb_s = sqrtf(bsx * bsx + bsy * bsy + bsz * bsz);
                float f_n  = nb_n / (nb_n + nb_s + EPSN);
                float vnx = (jnx - jsx) * f_n, vny = (jny - jsy) * f_n, vnz = (jnz - jsz) * f_n;
                float f_s  = -nb_s / (nb_n + EPSN);
                float vsx = vnx * f_s, vsy = vny * f_s, vsz = vnz * f_s;
                float n_n = sqrtf(vnx * vnx + vny * vny + vnz * vnz);
                float n_s = sqrtf(vsx * vsx + vsy * vsy + vsz * vsz);
                float tox = jnx - vnx, toy = jny - vny, toz = jnz - vnz;
                float vpx = lx - tox, vpy = ly - toy, vpz = lz - toz;
                float inn = 1.f / (n_n + EPSN), ins = 1.f / (n_s + EPSN);
                float r_n = (vpx * vnx + vpy * vny + vpz * vnz) * inn * inn;
                float r_s = (vpx * vsx + vpy * vsy + vpz * vsz) * ins * ins;
                float zt  = (r_n * a_n + be_n) - (r_s * a_s + be_s);
                float w   = 1.f / (1.f + __expf(-zt));
                float ax = -ntn(jpx * w), ay = -ntn(jpy * w), az = -ntn(jpz * w);
                // Rodrigues
                float ang = sqrtf(ax * ax + ay * ay + az * az);
                float den = (ang < 1e-12f) ? 1.f : ang;
                float ux = ax / den, uy = ay / den, uz = az / den;
                float ca = __cosf(ang), sa = __sinf(ang), C = 1.f - ca;
                float r00 = ca + ux * ux * C, r01 = ux * uy * C - uz * sa, r02 = ux * uz * C + uy * sa;
                float r10 = uy * ux * C + uz * sa, r11 = ca + uy * uy * C, r12 = uy * uz * C - ux * sa;
                float r20 = uz * ux * C - uy * sa, r21 = uz * uy * C + ux * sa, r22 = ca + uz * uz * C;
                float wx = lx - jcx, wy = ly - jcy, wz = lz - jcz;
                offx += ntn(r00 * wx + r01 * wy + r02 * wz + jcx - lx);
                offy += ntn(r10 * wx + r11 * wy + r12 * wz + jcy - ly);
                offz += ntn(r20 * wx + r21 * wy + r22 * wz + jcz - lz);
            }
            lx += offx; ly += offy; lz += offz;

            // x_in = [pts_local(3), c(4)], K padded to 32
            _Float16* row = act + lane * 128;
            row[0] = (_Float16)lx;    row[1] = (_Float16)ly;   row[2] = (_Float16)lz;
            row[3] = (_Float16)cj[0]; row[4] = (_Float16)cj[1];
            row[5] = (_Float16)cj[2]; row[6] = (_Float16)cj[3];
#pragma unroll
            for (int k = 7; k < 32; ++k) row[k] = (_Float16)0.f;
        }
        // ------------------ MLP: 7->128->128->128 (WMMA f16, f32 accum) -------
        gemm_layer<32, 1, true>(act, s_w0t, b0g, lane);
        gemm_layer<128, 4, true>(act, s_w1t, b1g, lane);
        gemm_layer<128, 4, true>(act, s_w2t, b2g, lane);

        // ------------------ final layer 128 -> 1 (N padded to 16) -------------
        v16h af[4];
#pragma unroll
        for (int kk = 0; kk < 4; ++kk) af[kk] = load_a_frag(act, lane, kk * 32);
        v8f acc = {};
#pragma unroll
        for (int kk = 0; kk < 4; ++kk) {
            v16h bf = load_b_frag(s_w3t, 128, lane, 0, kk * 32);
            acc = __builtin_amdgcn_wmma_f32_16x16x32_f16(
                false, af[kk], false, bf, (short)0, acc, false, false);
        }
        if ((lane & 15) == 0) {  // column N==0 lives in lanes 0 (rows 0-7) / 16 (rows 8-15)
            float* po = predws + j * NV + tile * 16;
#pragma unroll
            for (int g = 0; g < 8; ++g) {
                int M = (lane < 16) ? g : 8 + g;
                po[M] = acc[g] + b3v;
            }
        }
    }
}

// ---------------------------------------------------------------------------
// Kernel 3: softmin blend over the J=24 parts
// ---------------------------------------------------------------------------
__global__ void softmin_kernel(const float* __restrict__ predws, float* __restrict__ out) {
    int v = blockIdx.x * blockDim.x + threadIdx.x;
    if (v >= NV) return;
    float p[NJ];
    float minv = 3.402823466e38f;
#pragma unroll
    for (int j = 0; j < NJ; ++j) {
        p[j] = predws[j * NV + v];
        minv = fminf(minv, p[j]);
    }
    float s = 0.f, a = 0.f;
#pragma unroll
    for (int j = 0; j < NJ; ++j) {
        float d = p[j] - minv;
        float e = __expf(-200.f * d);
        s += e;
        a += d * e;
    }
    out[v] = a / s + minv;
}

// ---------------------------------------------------------------------------
extern "C" void kernel_launch(void* const* d_in, const int* in_sizes, int n_in,
                              void* d_out, int out_size, void* d_ws, size_t ws_size,
                              hipStream_t stream) {
    const float* pts    = (const float*)d_in[0];
    const float* Bcond  = (const float*)d_in[1];
    const float* Btr    = (const float*)d_in[2];
    const float* Brot   = (const float*)d_in[3];
    const float* Bneigh = (const float*)d_in[4];
    const float* ralpha = (const float*)d_in[5];
    const float* rbeta  = (const float*)d_in[6];
    const float* cW0 = (const float*)d_in[7];
    const float* cb0 = (const float*)d_in[8];
    const float* cW1 = (const float*)d_in[9];
    const float* cb1 = (const float*)d_in[10];
    const float* cW2 = (const float*)d_in[11];
    const float* cb2 = (const float*)d_in[12];
    const float* W0 = (const float*)d_in[13];
    const float* b0 = (const float*)d_in[14];
    const float* W1 = (const float*)d_in[15];
    const float* b1 = (const float*)d_in[16];
    const float* W2 = (const float*)d_in[17];
    const float* b2 = (const float*)d_in[18];
    const float* W3 = (const float*)d_in[19];
    const float* b3 = (const float*)d_in[20];

    float* cws    = (float*)d_ws;                       // 24*4 floats
    float* predws = (float*)((char*)d_ws + 1024);       // 24*20000 floats

    // allow >64KB dynamic LDS for the fused MLP kernel (host-side, capture-safe)
    (void)hipFuncSetAttribute((const void*)parts_kernel,
                              hipFuncAttributeMaxDynamicSharedMemorySize,
                              (int)SMEM_BYTES);

    cond_kernel<<<1, 32, 0, stream>>>(Bcond, cW0, cb0, cW1, cb1, cW2, cb2, cws);

    dim3 grid((NTILES + TILES_PER_BLK - 1) / TILES_PER_BLK, NJ);   // (20, 24)
    parts_kernel<<<grid, 256, SMEM_BYTES, stream>>>(pts, Btr, Brot, Bneigh, ralpha, rbeta,
                                                    W0, b0, W1, b1, W2, b2, W3, b3,
                                                    cws, predws);

    softmin_kernel<<<(NV + 255) / 256, 256, 0, stream>>>(predws, (float*)d_out);
}
